// GNN_44040594653300
// MI455X (gfx1250) — matrix operand order repaired and checked by reading, not compiled, not run on hardware
//
#include <hip/hip_runtime.h>
#include <math.h>

// ---------------------------------------------------------------------------
// Problem constants (from reference)
// ---------------------------------------------------------------------------
#define NNODES   24576      // W_DIM * BS = 384*64
#define CIN      672
#define NEDGES   393216
#define BATCH    64

// Workspace layout (bytes, all 256-aligned)
#define OFF_HB    0u                    // h bf16 (24576*672*2 = 33,030,144); reused later for h1 bf16
#define OFF_W1    33030144u             // W1 combined bf16 (256*672*2 = 344,064)
#define OFF_W2    33374208u             // W2 combined bf16 (128*128*2 = 32,768)
#define OFF_Y     33406976u             // Y (24576*256*4 = 25,165,824); reused for Y2 (24576*128*4)
#define OFF_ACC   58572800u             // acc (24576*128*4 = 12,582,912); reused for layer-2 acc
#define OFF_POOL  71155712u             // pooled (768*64*4 = 196,608)

// ---------------------------------------------------------------------------
// Types for WMMA
// ---------------------------------------------------------------------------
typedef __attribute__((ext_vector_type(16))) __bf16 bf16x16;
typedef __attribute__((ext_vector_type(8)))  __bf16 bf16x8;
typedef __attribute__((ext_vector_type(8)))  float  f32x8;

union AFrag { bf16x16 v; bf16x8 h[2]; };

// ---------------------------------------------------------------------------
// fp32 -> bf16 conversion kernels
// ---------------------------------------------------------------------------
__global__ void cvt_f32_to_bf16(const float* __restrict__ in, __bf16* __restrict__ out,
                                long long n) {
  long long i = (long long)blockIdx.x * blockDim.x + threadIdx.x;
  if (i < n) out[i] = (__bf16)in[i];
}

// pack two fp32 arrays back-to-back as one bf16 array (fused weight [W_rel; W_root])
__global__ void pack2_f32_to_bf16(const float* __restrict__ a, const float* __restrict__ b,
                                  __bf16* __restrict__ out, long long na, long long total) {
  long long i = (long long)blockIdx.x * blockDim.x + threadIdx.x;
  if (i < total) out[i] = (__bf16)((i < na) ? a[i] : b[i - na]);
}

// ---------------------------------------------------------------------------
// bf16 WMMA GEMM: C[M x N] (f32, row-major) = A[M x K] (bf16, rm) @ B[N x K]^T (bf16, rm)
// Block = 4 waves; each wave -> 16x64 tile; block -> 64x64 tile.
// 16-bit A 16x32 / B 32x16 use contiguous k (16*half + s), not the ISA table order; f32 C/D per §7.12.2.
// ---------------------------------------------------------------------------
__global__ void __launch_bounds__(128)
gemm_bf16_wmma(const __bf16* __restrict__ A, const __bf16* __restrict__ B,
               float* __restrict__ Cmat, int M, int N, int K) {
  const int lane = threadIdx.x & 31;
  const int wave = threadIdx.x >> 5;
  const int m0   = blockIdx.y * 64 + wave * 16;
  const int n0   = blockIdx.x * 64;
  const int half = lane >> 4;          // 0: lanes 0-15, 1: lanes 16-31
  const int l15  = lane & 15;

  f32x8 acc0 = {}, acc1 = {}, acc2 = {}, acc3 = {};

  // A lane base: row = m0 + (lane&15); 16 contiguous bf16 at K-offset half*16
  const __bf16* arow = A + (size_t)(m0 + l15) * K + half * 16;
  // B lane base: row n (weight row, since B[k][n] = W[n][k]); 16 contiguous bf16 at half*16
  const __bf16* brow0 = B + (size_t)(n0 +  0 + l15) * K + half * 16;
  const __bf16* brow1 = B + (size_t)(n0 + 16 + l15) * K + half * 16;
  const __bf16* brow2 = B + (size_t)(n0 + 32 + l15) * K + half * 16;
  const __bf16* brow3 = B + (size_t)(n0 + 48 + l15) * K + half * 16;

  for (int kk = 0; kk < K; kk += 32) {
    AFrag a;
    a.h[0] = *reinterpret_cast<const bf16x8*>(arow + kk);
    a.h[1] = *reinterpret_cast<const bf16x8*>(arow + kk + 8);
    bf16x16 fb0 = *reinterpret_cast<const bf16x16*>(brow0 + kk);
    bf16x16 fb1 = *reinterpret_cast<const bf16x16*>(brow1 + kk);
    bf16x16 fb2 = *reinterpret_cast<const bf16x16*>(brow2 + kk);
    bf16x16 fb3 = *reinterpret_cast<const bf16x16*>(brow3 + kk);

    __builtin_prefetch(arow + kk + 64, 0, 0);   // global_prefetch_b8 on gfx1250

    // 8 args: (neg_a, A, neg_b, B, c_mod, C, reuse_a, reuse_b)
    acc0 = __builtin_amdgcn_wmma_f32_16x16x32_bf16(false, a.v, false, fb0, (short)0, acc0, false, false);
    acc1 = __builtin_amdgcn_wmma_f32_16x16x32_bf16(false, a.v, false, fb1, (short)0, acc1, false, false);
    acc2 = __builtin_amdgcn_wmma_f32_16x16x32_bf16(false, a.v, false, fb2, (short)0, acc2, false, false);
    acc3 = __builtin_amdgcn_wmma_f32_16x16x32_bf16(false, a.v, false, fb3, (short)0, acc3, false, false);
  }

  // D layout: VGPR r -> rows r (lanes 0-15) / r+8 (lanes 16-31); col = n-tile + lane&15
  float* crow = Cmat + (size_t)(m0 + half * 8) * N + n0 + l15;
#pragma unroll
  for (int r = 0; r < 8; ++r) {
    crow[(size_t)r * N +  0] = acc0[r];
    crow[(size_t)r * N + 16] = acc1[r];
    crow[(size_t)r * N + 32] = acc2[r];
    crow[(size_t)r * N + 48] = acc3[r];
  }
}

// ---------------------------------------------------------------------------
// acc[i,f] = Y[i, off+f] + bias[f]   (root-lin contribution + bias, pre-scatter init)
// ---------------------------------------------------------------------------
__global__ void add_root_bias(const float* __restrict__ Y, int ystride, int off,
                              const float* __restrict__ bias, float* __restrict__ acc,
                              int F, long long total) {
  long long t = (long long)blockIdx.x * blockDim.x + threadIdx.x;
  if (t >= total) return;
  long long i = t / F;
  int f = (int)(t - i * F);
  acc[t] = Y[i * ystride + off + f] + bias[f];
}

// ---------------------------------------------------------------------------
// Edge scatter: acc[dst, f] += Y[src, f] * edge_attr[e]   (rel part lives in cols [0,F))
// ---------------------------------------------------------------------------
__global__ void scatter_edges(const float* __restrict__ Y, int ystride,
                              const int* __restrict__ ei, const float* __restrict__ ea,
                              float* __restrict__ acc, int F, long long total) {
  long long t = (long long)blockIdx.x * blockDim.x + threadIdx.x;
  if (t >= total) return;
  long long e = t / F;
  int f = (int)(t - e * F);
  int s = ei[e];
  int d = ei[NEDGES + e];
  atomicAdd(&acc[(size_t)d * F + f], Y[(size_t)s * ystride + f] * ea[e]);
}

// ---------------------------------------------------------------------------
// h_next_bf16 = bf16(tanh(acc))
// ---------------------------------------------------------------------------
__global__ void tanh_to_bf16(const float* __restrict__ acc, __bf16* __restrict__ out,
                             long long n) {
  long long i = (long long)blockIdx.x * blockDim.x + threadIdx.x;
  if (i < n) out[i] = (__bf16)tanhf(acc[i]);
}

// ---------------------------------------------------------------------------
// pooled[g,f] = mean over 32 nodes of tanh(acc2[g*32+r, f])   (768 groups, 64 feats)
// ---------------------------------------------------------------------------
__global__ void tanh_pool_kernel(const float* __restrict__ acc, float* __restrict__ pooled) {
  int t = blockIdx.x * blockDim.x + threadIdx.x;
  if (t >= 768 * 64) return;
  int g = t >> 6, f = t & 63;
  const float* p = acc + (size_t)g * 32 * 64 + f;
  float s = 0.f;
#pragma unroll
  for (int r = 0; r < 32; ++r) s += tanhf(p[r * 64]);
  pooled[t] = s * (1.0f / 32.0f);
}

// ---------------------------------------------------------------------------
// Head: conv1d x3 + MLP, one block (64 threads) per batch element, LDS staging
// ---------------------------------------------------------------------------
__global__ void __launch_bounds__(64)
head_kernel(const float* __restrict__ pooled,
            const float* __restrict__ cw1, const float* __restrict__ cb1,
            const float* __restrict__ cw2, const float* __restrict__ cb2,
            const float* __restrict__ cw3, const float* __restrict__ cb3,
            const float* __restrict__ m1w, const float* __restrict__ m1b,
            const float* __restrict__ m2w, const float* __restrict__ m2b,
            const float* __restrict__ m3w, const float* __restrict__ m3b,
            float* __restrict__ out) {
  __shared__ float g[12 * 64];
  __shared__ float o1[8 * 63];
  __shared__ float o2[4 * 62];
  __shared__ float o3[61];
  __shared__ float v1[16];
  __shared__ float v2[8];

  const int b = blockIdx.x;
  const int t = threadIdx.x;

  for (int i = t; i < 12 * 64; i += 64) {
    int c = i >> 6, l = i & 63;
    g[i] = pooled[((size_t)b * 12 + c) * 64 + l];
  }
  __syncthreads();

  for (int i = t; i < 8 * 63; i += 64) {
    int co = i / 63, l = i % 63;
    float s = cb1[co];
#pragma unroll
    for (int ci = 0; ci < 12; ++ci)
      s += cw1[co * 24 + ci * 2] * g[ci * 64 + l] + cw1[co * 24 + ci * 2 + 1] * g[ci * 64 + l + 1];
    o1[i] = fmaxf(s, 0.f);
  }
  __syncthreads();

  for (int i = t; i < 4 * 62; i += 64) {
    int co = i / 62, l = i % 62;
    float s = cb2[co];
#pragma unroll
    for (int ci = 0; ci < 8; ++ci)
      s += cw2[co * 16 + ci * 2] * o1[ci * 63 + l] + cw2[co * 16 + ci * 2 + 1] * o1[ci * 63 + l + 1];
    o2[i] = fmaxf(s, 0.f);
  }
  __syncthreads();

  for (int l = t; l < 61; l += 64) {
    float s = cb3[0];
#pragma unroll
    for (int ci = 0; ci < 4; ++ci)
      s += cw3[ci * 2] * o2[ci * 62 + l] + cw3[ci * 2 + 1] * o2[ci * 62 + l + 1];
    o3[l] = fmaxf(s, 0.f);
  }
  __syncthreads();

  if (t < 16) {
    float s = m1b[t];
    for (int i = 0; i < 61; ++i) s += m1w[t * 61 + i] * o3[i];
    v1[t] = fmaxf(s, 0.f);
  }
  __syncthreads();
  if (t < 8) {
    float s = m2b[t];
#pragma unroll
    for (int i = 0; i < 16; ++i) s += m2w[t * 16 + i] * v1[i];
    v2[t] = fmaxf(s, 0.f);
  }
  __syncthreads();
  if (t == 0) {
    float s = m3b[0];
#pragma unroll
    for (int i = 0; i < 8; ++i) s += m3w[i] * v2[i];
    out[b] = s;
  }
}

// ---------------------------------------------------------------------------
// Launch
// ---------------------------------------------------------------------------
static inline long long divUpLL(long long a, long long b) { return (a + b - 1) / b; }

extern "C" void kernel_launch(void* const* d_in, const int* in_sizes, int n_in,
                              void* d_out, int out_size, void* d_ws, size_t ws_size,
                              hipStream_t stream) {
  const float* x        = (const float*)d_in[0];
  const float* edge_attr= (const float*)d_in[1];
  const float* w1_rel   = (const float*)d_in[2];
  const float* b1_rel   = (const float*)d_in[3];
  const float* w1_root  = (const float*)d_in[4];
  const float* w2_rel   = (const float*)d_in[5];
  const float* b2_rel   = (const float*)d_in[6];
  const float* w2_root  = (const float*)d_in[7];
  const float* cw1 = (const float*)d_in[8];  const float* cb1 = (const float*)d_in[9];
  const float* cw2 = (const float*)d_in[10]; const float* cb2 = (const float*)d_in[11];
  const float* cw3 = (const float*)d_in[12]; const float* cb3 = (const float*)d_in[13];
  const float* m1w = (const float*)d_in[14]; const float* m1b = (const float*)d_in[15];
  const float* m2w = (const float*)d_in[16]; const float* m2b = (const float*)d_in[17];
  const float* m3w = (const float*)d_in[18]; const float* m3b = (const float*)d_in[19];
  const int*   eidx = (const int*)d_in[20];
  float* out = (float*)d_out;

  char* ws = (char*)d_ws;
  __bf16* hB   = (__bf16*)(ws + OFF_HB);     // h (24576x672) bf16
  __bf16* h1B  = (__bf16*)(ws + OFF_HB);     // reused: h1 (24576x128) bf16
  __bf16* W1B  = (__bf16*)(ws + OFF_W1);     // [w1_rel; w1_root] (256x672) bf16
  __bf16* W2B  = (__bf16*)(ws + OFF_W2);     // [w2_rel; w2_root] (128x128) bf16
  float*  Y    = (float*)(ws + OFF_Y);       // Y1 (24576x256), then Y2 (24576x128)
  float*  acc  = (float*)(ws + OFF_ACC);     // h1 acc (24576x128), then h2 acc (24576x64)
  float*  pool = (float*)(ws + OFF_POOL);    // pooled (768x64)

  // --- convert inputs to bf16 ---
  {
    long long n = (long long)NNODES * CIN;
    cvt_f32_to_bf16<<<(unsigned)divUpLL(n, 256), 256, 0, stream>>>(x, hB, n);
    long long na1 = 128LL * CIN, t1 = 256LL * CIN;
    pack2_f32_to_bf16<<<(unsigned)divUpLL(t1, 256), 256, 0, stream>>>(w1_rel, w1_root, W1B, na1, t1);
    long long na2 = 64LL * 128, t2 = 128LL * 128;
    pack2_f32_to_bf16<<<(unsigned)divUpLL(t2, 256), 256, 0, stream>>>(w2_rel, w2_root, W2B, na2, t2);
  }

  // --- Layer 1: Y1 = h @ [w1_rel; w1_root]^T  (24576 x 256, K=672) ---
  gemm_bf16_wmma<<<dim3(256 / 64, NNODES / 64), 128, 0, stream>>>(hB, W1B, Y, NNODES, 256, CIN);
  {
    long long t = (long long)NNODES * 128;
    add_root_bias<<<(unsigned)divUpLL(t, 256), 256, 0, stream>>>(Y, 256, 128, b1_rel, acc, 128, t);
    long long te = (long long)NEDGES * 128;
    scatter_edges<<<(unsigned)divUpLL(te, 256), 256, 0, stream>>>(Y, 256, eidx, edge_attr, acc, 128, te);
    tanh_to_bf16<<<(unsigned)divUpLL(t, 256), 256, 0, stream>>>(acc, h1B, t);
  }

  // --- Layer 2: Y2 = h1 @ [w2_rel; w2_root]^T  (24576 x 128, K=128) ---
  gemm_bf16_wmma<<<dim3(128 / 64, NNODES / 64), 128, 0, stream>>>(h1B, W2B, Y, NNODES, 128, 128);
  {
    long long t = (long long)NNODES * 64;
    add_root_bias<<<(unsigned)divUpLL(t, 256), 256, 0, stream>>>(Y, 128, 64, b2_rel, acc, 64, t);
    long long te = (long long)NEDGES * 64;
    scatter_edges<<<(unsigned)divUpLL(te, 256), 256, 0, stream>>>(Y, 128, eidx, edge_attr, acc, 64, te);
  }

  // --- tanh + global mean pool (768 groups of 32 nodes, 64 feats) ---
  tanh_pool_kernel<<<(768 * 64) / 256, 256, 0, stream>>>(acc, pool);

  // --- conv/MLP head, one block per batch element ---
  head_kernel<<<BATCH, 64, 0, stream>>>(pool, cw1, cb1, cw2, cb2, cw3, cb3,
                                        m1w, m1b, m2w, m2b, m3w, m3b, out);

  (void)in_sizes; (void)n_in; (void)out_size; (void)ws_size;
}
